// ClusterMerging_43602507989796
// MI455X (gfx1250) — matrix operand order repaired and checked by reading, not compiled, not run on hardware
//
#include <hip/hip_runtime.h>

// ---------------- problem constants (from reference) ----------------
#define B_      8
#define N_      4096
#define C_      384
#define H_      12
#define D_      3
#define Kc_     256          // clusters per batch
#define M_      52           // members per cluster
#define Z_      (B_ * Kc_)   // 2048
#define MQ_     10           // len(range(2,52,5))
#define START_  2
#define SKIP_   5
#define FOURC_  (4 * C_)     // 1536
#define TWOC_   (2 * C_)     // 768
#define CH_     (C_ / H_)    // 32 per-head dim
#define ROWS_   (B_ * N_)    // 32768 GEMM M

typedef __attribute__((ext_vector_type(16))) __bf16 v16bf;
typedef __attribute__((ext_vector_type(8)))  __bf16 v8bf;
typedef __attribute__((ext_vector_type(8)))  float  v8f;
typedef __attribute__((ext_vector_type(4)))  int    v4i;

union V16U { v16bf v; v8bf h[2]; uint4 q[2]; };

#define AS1 __attribute__((address_space(1)))
#define AS3 __attribute__((address_space(3)))

#if defined(__has_builtin)
#if __has_builtin(__builtin_amdgcn_global_load_async_to_lds_b128)
#define HAVE_ASYNC_LDS 1
#endif
#if __has_builtin(__builtin_amdgcn_s_wait_asynccnt)
#define HAVE_WAIT_ASYNC 1
#endif
#endif

// ---------------- 1) LayerNorm -> bf16 ----------------
__global__ __launch_bounds__(128)
void cm_layernorm_bf16(const float* __restrict__ feat,
                       const float* __restrict__ w,
                       const float* __restrict__ b,
                       __bf16* __restrict__ out) {
  int row = blockIdx.x;
  const float* f = feat + (size_t)row * C_;
  int tid = threadIdx.x;
  float s = 0.f, ss = 0.f;
  for (int i = tid; i < C_; i += 128) { float v = f[i]; s += v; ss += v * v; }
  __shared__ float sh0[128], sh1[128];
  sh0[tid] = s; sh1[tid] = ss; __syncthreads();
  for (int o = 64; o > 0; o >>= 1) {
    if (tid < o) { sh0[tid] += sh0[tid + o]; sh1[tid] += sh1[tid + o]; }
    __syncthreads();
  }
  float mu   = sh0[0] * (1.0f / C_);
  float var  = sh1[0] * (1.0f / C_) - mu * mu;
  float rstd = rsqrtf(var + 1e-5f);
  __bf16* o = out + (size_t)row * C_;
  for (int i = tid; i < C_; i += 128)
    o[i] = (__bf16)((f[i] - mu) * rstd * w[i] + b[i]);
}

// ---------------- weight fp32 -> bf16, transposed: Wt[n][k] ----------------
__global__ __launch_bounds__(256)
void cm_transpose_w_bf16(const float* __restrict__ in,   // [C_][FOURC_]
                         __bf16* __restrict__ out) {     // [FOURC_][C_]
  int i = blockIdx.x * blockDim.x + threadIdx.x;
  if (i >= C_ * FOURC_) return;
  int k = i / FOURC_, n = i % FOURC_;
  out[(size_t)n * C_ + k] = (__bf16)in[i];
}

// ---------------- 2) WMMA bf16 GEMM: qkv = x @ W + b ----------------
// block: 256 threads = 8 waves (4 M-waves x 2 N-waves); tile 128x128, BK=32.
// A staged in double-buffered LDS via async-to-LDS; B fragments loaded straight
// from the pre-transposed (L2-resident) weight matrix via 4 persistent,
// increment-only pointers. K loop fully unrolled for register rotation.
#define BM 128
#define BN 128
#define BK 32
#define LDP 40   // padded LDS stride in halfs (80 B, 16B aligned)

__global__ __launch_bounds__(256)
void cm_qkv_gemm(const __bf16* __restrict__ A,    // [ROWS_][C_]
                 const __bf16* __restrict__ Wt,   // [FOURC_][C_] (transposed W)
                 const float*  __restrict__ bias,
                 float* __restrict__ Cout) {
  __shared__ __bf16 As[2][BM][LDP];
  const int Ndim = FOURC_;
  int tid  = threadIdx.x;
  int wave = tid >> 5;
  int lane = tid & 31;
  int mw = wave & 3;     // 0..3
  int nw = wave >> 2;    // 0..1
  int lhalf = lane >> 4; // 0/1
  int lmod  = lane & 15;
  int tileM = blockIdx.x * BM;
  int tileN = blockIdx.y * BN;

  v8f acc[2][4];
  v8f zero = {0.f,0.f,0.f,0.f,0.f,0.f,0.f,0.f};
  #pragma unroll
  for (int i = 0; i < 2; i++)
    #pragma unroll
    for (int j = 0; j < 4; j++) acc[i][j] = zero;

  int ar  = tid >> 1;              // A stage row 0..127
  int acb = (tid & 1) * 16;        // 0 or 16 (halfs)

  // persistent per-lane B pointers (advance by BK halfs = 4 uint4 per K-step)
  const uint4* bptr[4];
  #pragma unroll
  for (int ni = 0; ni < 4; ni++) {
    int coln = tileN + nw * 64 + ni * 16 + lmod;
    bptr[ni] = (const uint4*)(Wt + (size_t)coln * C_ + lhalf * 16);
  }

  auto stage = [&](int buf, int k0) {
    const __bf16* src = A + (size_t)(tileM + ar) * C_ + (k0 + acb);
#if defined(HAVE_ASYNC_LDS)
    __builtin_amdgcn_global_load_async_to_lds_b128(
        (AS1 v4i*)src,       (AS3 v4i*)&As[buf][ar][acb],     0, 0);
    __builtin_amdgcn_global_load_async_to_lds_b128(
        (AS1 v4i*)(src + 8), (AS3 v4i*)&As[buf][ar][acb + 8], 0, 0);
#else
    const uint4* s4 = (const uint4*)src;
    *(uint4*)&As[buf][ar][acb]     = s4[0];
    *(uint4*)&As[buf][ar][acb + 8] = s4[1];
#endif
  };

  int buf = 0;
  stage(0, 0);

  #pragma unroll
  for (int k0 = 0; k0 < C_; k0 += BK) {
#if defined(HAVE_ASYNC_LDS)
#if defined(HAVE_WAIT_ASYNC)
    __builtin_amdgcn_s_wait_asynccnt(0);
#else
    asm volatile("s_wait_asynccnt 0x0" ::: "memory");
#endif
#endif
    __syncthreads();                       // publish As[buf] to all waves
    if (k0 + BK < C_) {
      stage(buf ^ 1, k0 + BK);             // overlap next-tile staging
      // pull next B K-segment toward L2/L0 (gfx1250 global_prefetch_b8)
      __builtin_prefetch(Wt + (size_t)(tileN + nw * 64 + lmod) * C_ + k0 + BK, 0, 0);
    }

    // --- batch-load all 4 B fragments (distinct regs, loads overlap) ---
    V16U bfr[4];
    #pragma unroll
    for (int ni = 0; ni < 4; ni++) {
      bfr[ni].q[0] = bptr[ni][0];
      bfr[ni].q[1] = bptr[ni][1];
      bptr[ni] += 4;                       // advance BK halfs (64 B)
    }
    // --- A fragments for this wave's 2 M-subtiles (contiguous k = 16*lhalf + s, matching B) ---
    V16U af[2];
    #pragma unroll
    for (int mi = 0; mi < 2; mi++) {
      int row = mw * 32 + mi * 16 + lmod;
      af[mi].q[0] = *(const uint4*)&As[buf][row][lhalf * 16];
      af[mi].q[1] = *(const uint4*)&As[buf][row][lhalf * 16 + 8];
    }
    // --- 8 back-to-back WMMAs ---
    #pragma unroll
    for (int ni = 0; ni < 4; ni++) {
      #pragma unroll
      for (int mi = 0; mi < 2; mi++) {
        acc[mi][ni] = __builtin_amdgcn_wmma_f32_16x16x32_bf16(
            false, af[mi].v, false, bfr[ni].v, (short)0, acc[mi][ni], false, false);
      }
    }
    buf ^= 1;
  }

  // --- store: D element (m = r + 8*lhalf, n = lmod) per ISA C/D layout ---
  #pragma unroll
  for (int mi = 0; mi < 2; mi++) {
    #pragma unroll
    for (int ni = 0; ni < 4; ni++) {
      int colg = tileN + nw * 64 + ni * 16 + lmod;
      float bv = bias[colg];
      #pragma unroll
      for (int r = 0; r < 8; r++) {
        int rowg = tileM + mw * 32 + mi * 16 + lhalf * 8 + r;
        Cout[(size_t)rowg * Ndim + colg] = acc[mi][ni][r] + bv;
      }
    }
  }
}

// ---------------- 3) gather + per-(z,h) attention ----------------
__global__ __launch_bounds__(64)
void cm_attention(const float* __restrict__ qkv,  // [B_*N_][1536]
                  const float* __restrict__ pos,  // [B_*N_][3]
                  const int*   __restrict__ member_idx, // [Z_][M_]
                  const float* __restrict__ pos_w, // [3][H_]
                  const float* __restrict__ pos_b, // [H_]
                  float* __restrict__ feat_c)      // [Z_][MQ_][768]
{
  int zh = blockIdx.x;
  int z = zh / H_, h = zh % H_;
  int bb = z / Kc_;
  int tid = threadIdx.x;

  __shared__ float Ksh[M_][CH_ + 1];
  __shared__ float Vsh[M_][2 * CH_];
  __shared__ float Qsh[MQ_][CH_ + 1];
  __shared__ float Pg[M_][3];
  __shared__ float Att[MQ_][M_];
  __shared__ int   mem[M_];

  for (int j = tid; j < M_; j += 64) mem[j] = member_idx[(size_t)z * M_ + j];
  __syncthreads();

  for (int t = tid; t < M_ * CH_; t += 64) {
    int j = t >> 5, c = t & 31;
    size_t base = ((size_t)bb * N_ + mem[j]) * FOURC_;
    Ksh[j][c]       = qkv[base + C_     + h * CH_ + c];
    Vsh[j][c]       = qkv[base + 2 * C_ + h * CH_ + c];
    Vsh[j][CH_ + c] = qkv[base + 3 * C_ + h * CH_ + c];
  }
  for (int t = tid; t < M_ * 3; t += 64) {
    int j = t / 3, c = t % 3;
    Pg[j][c] = pos[((size_t)bb * N_ + mem[j]) * 3 + c];
  }
  for (int t = tid; t < MQ_ * CH_; t += 64) {
    int i = t >> 5, c = t & 31;
    int j = START_ + i * SKIP_;
    size_t base = ((size_t)bb * N_ + mem[j]) * FOURC_;
    Qsh[i][c] = qkv[base + h * CH_ + c];
  }
  __syncthreads();

  if (tid < M_) {
    float s = 0.f;
    for (int c = 0; c < CH_; c++) s += Ksh[tid][c] * Ksh[tid][c];
    float inv = 1.0f / (sqrtf(s) + 1e-8f);
    for (int c = 0; c < CH_; c++) Ksh[tid][c] *= inv;
  }
  if (tid < MQ_) {
    float s = 0.f;
    for (int c = 0; c < CH_; c++) s += Qsh[tid][c] * Qsh[tid][c];
    float inv = 1.0f / (sqrtf(s) + 1e-8f);
    for (int c = 0; c < CH_; c++) Qsh[tid][c] *= inv;
  }
  __syncthreads();

  for (int t = tid; t < MQ_ * M_; t += 64) {
    int i = t / M_, j = t % M_;
    float s = 0.f;
    for (int c = 0; c < CH_; c++) s += Qsh[i][c] * Ksh[j][c];
    int jd = START_ + i * SKIP_;
    float bsum = pos_b[h];
    for (int d = 0; d < 3; d++) bsum += (Pg[j][d] - Pg[jd][d]) * pos_w[d * H_ + h];
    Att[i][j] = s + bsum;
  }
  __syncthreads();

  for (int t = tid; t < MQ_ * 2 * CH_; t += 64) {
    int i = t >> 6, c = t & 63;
    float s = 0.f;
    for (int j = 0; j < M_; j++) s += Att[i][j] * Vsh[j][c];
    feat_c[((size_t)z * MQ_ + i) * TWOC_ + h * 2 * CH_ + c] = s;
  }
}

// ---------------- 4) merge: presence -> scan -> scatter ----------------
__global__ __launch_bounds__(256)
void cm_mark_present(const int* __restrict__ member_idx, int* __restrict__ present) {
  int idx = blockIdx.x * blockDim.x + threadIdx.x;
  if (idx >= Z_ * MQ_) return;
  int z = idx / MQ_, t = idx % MQ_;
  int b = z / Kc_;
  int v = member_idx[(size_t)z * M_ + START_ + t * SKIP_];
  present[b * N_ + v] = 1;
}

__global__ __launch_bounds__(256)
void cm_scan_present(const int* __restrict__ present, int* __restrict__ psum) {
  int b = blockIdx.x;
  const int* pr = present + b * N_;
  int* ps = psum + b * N_;
  int tid = threadIdx.x;
  __shared__ int part[256], excl[256];
  int base = tid * 16;
  int loc[16], s = 0;
  for (int i = 0; i < 16; i++) { loc[i] = s; s += pr[base + i]; }
  part[tid] = s; __syncthreads();
  if (tid == 0) { int a = 0; for (int i = 0; i < 256; i++) { excl[i] = a; a += part[i]; } }
  __syncthreads();
  int off = excl[tid];
  for (int i = 0; i < 16; i++) ps[base + i] = off + loc[i];
}

__global__ __launch_bounds__(256)
void cm_scatter(const int* __restrict__ member_idx, const int* __restrict__ psum,
                const float* __restrict__ feat_c, const float* __restrict__ pos,
                float* __restrict__ pos_out, float* __restrict__ feat_out,
                float* __restrict__ counts, int n_new) {
  int e = blockIdx.x;                 // 0 .. B_*Kc_*MQ_-1
  int b  = e / (Kc_ * MQ_);
  int jj = e % (Kc_ * MQ_);
  int z = b * Kc_ + jj / MQ_;
  int t = jj % MQ_;
  __shared__ int sid, sv;
  if (threadIdx.x == 0) {
    int v = member_idx[(size_t)z * M_ + START_ + t * SKIP_];
    sv = v;
    sid = psum[b * N_ + v];
  }
  __syncthreads();
  int id = sid, v = sv;
  if (threadIdx.x == 0) atomicAdd(&counts[b * n_new + id], 1.0f);
  if (threadIdx.x < 3)
    pos_out[((size_t)b * n_new + id) * 3 + threadIdx.x] =
        pos[((size_t)b * N_ + v) * 3 + threadIdx.x];
  const float* src = feat_c + ((size_t)z * MQ_ + t) * TWOC_;
  float* dst = feat_out + ((size_t)b * n_new + id) * TWOC_;
  for (int c = threadIdx.x; c < TWOC_; c += 256) atomicAdd(&dst[c], src[c]);
}

__global__ __launch_bounds__(256)
void cm_finalize_feat(float* __restrict__ feat_out, const float* __restrict__ counts,
                      int n_new) {
  size_t total = (size_t)B_ * n_new * TWOC_;
  for (size_t i = (size_t)blockIdx.x * blockDim.x + threadIdx.x; i < total;
       i += (size_t)gridDim.x * blockDim.x) {
    float cnt = counts[i / TWOC_];
    feat_out[i] = feat_out[i] / fmaxf(cnt, 1.0f);
  }
}

// ---------------- 5) pos global mean/std normalize ----------------
__global__ __launch_bounds__(256)
void cm_pos_reduce(const float* __restrict__ pos_out, float* __restrict__ stats,
                   int n_new) {
  int total = B_ * n_new * 3;
  float s = 0.f, ss = 0.f;
  for (int i = blockIdx.x * blockDim.x + threadIdx.x; i < total;
       i += gridDim.x * blockDim.x) { float v = pos_out[i]; s += v; ss += v * v; }
  __shared__ float sh0[256], sh1[256];
  int tid = threadIdx.x;
  sh0[tid] = s; sh1[tid] = ss; __syncthreads();
  for (int o = 128; o > 0; o >>= 1) {
    if (tid < o) { sh0[tid] += sh0[tid + o]; sh1[tid] += sh1[tid + o]; }
    __syncthreads();
  }
  if (tid == 0) { atomicAdd(&stats[0], sh0[0]); atomicAdd(&stats[1], sh1[0]); }
}

__global__ __launch_bounds__(256)
void cm_pos_normalize(float* __restrict__ pos_out, const float* __restrict__ stats,
                      int n_new) {
  int total = B_ * n_new * 3;
  float inv  = 1.0f / (float)total;
  float mean = stats[0] * inv;
  float var  = stats[1] * inv - mean * mean;
  float ps   = sqrtf(var + 1e-5f);
  float rps  = 1.0f / ps;
  for (int i = blockIdx.x * blockDim.x + threadIdx.x; i < total;
       i += gridDim.x * blockDim.x)
    pos_out[i] = (pos_out[i] - mean) * rps;
}

// ---------------- host launcher ----------------
extern "C" void kernel_launch(void* const* d_in, const int* in_sizes, int n_in,
                              void* d_out, int out_size, void* d_ws, size_t ws_size,
                              hipStream_t stream) {
  const float* pos        = (const float*)d_in[0];
  const float* feat       = (const float*)d_in[1];
  const int*   member_idx = (const int*)d_in[2];
  // d_in[3] = batch_idx (derivable as z/Kc_), d_in[4] = k scalar — unused
  const float* norm_w = (const float*)d_in[5];
  const float* norm_b = (const float*)d_in[6];
  const float* qkv_w  = (const float*)d_in[7];
  const float* qkv_b  = (const float*)d_in[8];
  const float* pos_w  = (const float*)d_in[9];
  const float* pos_b  = (const float*)d_in[10];

  // outputs: pos_out (B, n_new, 3) then feat_out (B, n_new, 768); mask is None
  int n_new = out_size / (B_ * (D_ + TWOC_));
  float* pos_out  = (float*)d_out;
  float* feat_out = (float*)d_out + (size_t)B_ * n_new * D_;

  // workspace carve-up
  char* w = (char*)d_ws;
  __bf16* xbf    = (__bf16*)w; w += (size_t)ROWS_ * C_ * 2;
  __bf16* wtbf   = (__bf16*)w; w += (size_t)C_ * FOURC_ * 2;
  float*  qkv    = (float*)w;  w += (size_t)ROWS_ * FOURC_ * 4;
  float*  feat_c = (float*)w;  w += (size_t)Z_ * MQ_ * TWOC_ * 4;
  int*    present= (int*)w;    w += (size_t)B_ * N_ * 4;
  int*    psum   = (int*)w;    w += (size_t)B_ * N_ * 4;
  float*  counts = (float*)w;  w += (size_t)B_ * N_ * 4;
  float*  stats  = (float*)w;  w += 256;

  // zero-init accumulators / outputs (d_out & d_ws are poisoned)
  (void)hipMemsetAsync(d_out, 0, (size_t)out_size * sizeof(float), stream);
  (void)hipMemsetAsync(present, 0, (size_t)B_ * N_ * 4, stream);
  (void)hipMemsetAsync(counts, 0, (size_t)B_ * N_ * 4, stream);
  (void)hipMemsetAsync(stats, 0, 256, stream);

  // 1) weights -> bf16 transposed ; layernorm -> bf16
  cm_transpose_w_bf16<<<(C_ * FOURC_ + 255) / 256, 256, 0, stream>>>(qkv_w, wtbf);
  cm_layernorm_bf16<<<ROWS_, 128, 0, stream>>>(feat, norm_w, norm_b, xbf);

  // 2) WMMA GEMM: qkv = x @ W + b
  dim3 ggrid(ROWS_ / BM, FOURC_ / BN);
  cm_qkv_gemm<<<ggrid, 256, 0, stream>>>(xbf, wtbf, qkv_b, qkv);

  // 3) gather + attention
  cm_attention<<<Z_ * H_, 64, 0, stream>>>(qkv, pos, member_idx, pos_w, pos_b, feat_c);

  // 4) merge
  cm_mark_present<<<(Z_ * MQ_ + 255) / 256, 256, 0, stream>>>(member_idx, present);
  cm_scan_present<<<B_, 256, 0, stream>>>(present, psum);
  cm_scatter<<<B_ * Kc_ * MQ_, 256, 0, stream>>>(member_idx, psum, feat_c, pos,
                                                 pos_out, feat_out, counts, n_new);
  cm_finalize_feat<<<1024, 256, 0, stream>>>(feat_out, counts, n_new);

  // 5) pos normalize
  cm_pos_reduce<<<256, 256, 0, stream>>>(pos_out, stats, n_new);
  cm_pos_normalize<<<256, 256, 0, stream>>>(pos_out, stats, n_new);
}
